// NODEWrapper_with_ratio_tvf_rl_54657753809069
// MI455X (gfx1250) — matrix operand order repaired and checked by reading, not compiled, not run on hardware
//
#include <hip/hip_runtime.h>
#include <hip/hip_bf16.h>

typedef __bf16 bf16;
typedef __attribute__((ext_vector_type(16))) bf16  v16bf;
typedef __attribute__((ext_vector_type(8)))  float v8f;
typedef __attribute__((ext_vector_type(4)))  unsigned int u32x4;

#define KT1 3   /* 96/32  : layer-1 K tiles (81 padded to 96) */
#define KT2 8   /* 256/32 : layer-2/3 K tiles */

// fragment array sizes (elements)
#define W1F_ELEMS (16 * KT1 * 32 * 16)   /* 24576 */
#define W2F_ELEMS (16 * KT2 * 32 * 16)   /* 65536 */
#define W3F_ELEMS ( 8 * KT2 * 32 * 16)   /* 32768 */
#define NFRAG1 (16 * KT1 * 32)           /* 1536 */
#define NFRAG2 (16 * KT2 * 32)           /* 4096 */
#define NFRAG3 ( 8 * KT2 * 32)           /* 2048 */

#define MROWS 32                          /* batch rows per block (2 M-tiles) */

__device__ __forceinline__ float siluf(float x) {
  float s = 1.0f / (1.0f + __expf(-x));
  return x * s;
}
__device__ __forceinline__ float silup(float x) {
  float s = 1.0f / (1.0f + __expf(-x));
  return s * (1.0f + x * (1.0f - s));
}

// ---------------------------------------------------------------------------
// Pre-pass: f32 weights -> bf16 WMMA B-fragments, pre-swizzled.
// Fragment (kt, nt, lane) at ((nt*KT + kt)*32 + lane)*16 elements.
// B 32x16 bf16: lane L -> N = nt*16 + (L&15); element e -> K = kt*32 + 16*(L>>4) + e.
// ---------------------------------------------------------------------------
__global__ void prep_weights(const float* __restrict__ W1,
                             const float* __restrict__ W2,
                             const float* __restrict__ W3,
                             bf16* __restrict__ ws) {
  int f = blockIdx.x * blockDim.x + threadIdx.x;
  const float* W;
  int KT, Krows, Ncols, fl;
  bf16* dst;
  if (f < NFRAG1) {
    W = W1; KT = KT1; Krows = 81; Ncols = 256; dst = ws; fl = f;
  } else if (f < NFRAG1 + NFRAG2) {
    W = W2; KT = KT2; Krows = 256; Ncols = 256; dst = ws + W1F_ELEMS; fl = f - NFRAG1;
  } else if (f < NFRAG1 + NFRAG2 + NFRAG3) {
    W = W3; KT = KT2; Krows = 256; Ncols = 128; dst = ws + W1F_ELEMS + W2F_ELEMS; fl = f - NFRAG1 - NFRAG2;
  } else {
    return;
  }
  int lane = fl & 31;
  int tt   = fl >> 5;
  int kt   = tt % KT;
  int nt   = tt / KT;
  int n    = nt * 16 + (lane & 15);
  int half = lane >> 4;
  bf16* o  = dst + (size_t)fl * 16;
  for (int e = 0; e < 16; ++e) {
    int k = kt * 32 + 16 * half + e;
    float v = (k < Krows) ? W[(size_t)k * Ncols + n] : 0.0f;
    o[e] = (bf16)v;
  }
}

// ---------------------------------------------------------------------------
// Main kernel: 32 batch rows per block, 256 threads = 8 wave32s.
// Only W2 (the dominant 7-pass GEMM operand) is register-resident:
// 16 fragments = 128 VGPRs, keeping the wave under the 256-VGPR window
// (no s_set_vgpr_msb paging) at 4 waves/EU.
// ---------------------------------------------------------------------------
__launch_bounds__(256, 4)
__global__ void fm_divergence_kernel(const float* __restrict__ t,
                                     const float* __restrict__ x,
                                     const float* __restrict__ eps,
                                     const float* __restrict__ cond_u,
                                     const float* __restrict__ cond_v,
                                     const float* __restrict__ cond_f,
                                     const float* __restrict__ b1,
                                     const float* __restrict__ b2,
                                     const float* __restrict__ b3,
                                     const bf16* __restrict__ W1F,
                                     const bf16* __restrict__ W2F,
                                     const bf16* __restrict__ W3F,
                                     float* __restrict__ out) {
  __shared__ __align__(16) bf16  sIn[MROWS * 96];    // current variant input
  __shared__ __align__(16) bf16  sH1[MROWS * 256];   // h1 of current variant
  __shared__ __align__(16) bf16  sH2[MROWS * 256];   // h2 of current variant
  __shared__ __align__(16) bf16  sDZ1[MROWS * 256];  // tangent dz1 (persistent)
  __shared__ __align__(16) bf16  sDH[MROWS * 256];   // dh1 then dh2 of a/b variant
  __shared__ __align__(16) bf16  sDZ2[MROWS * 256];  // tangent dz2
  __shared__ float sE[MROWS * 64];                   // eps tile (f32)
  __shared__ float sOutF[MROWS * 64];                // ft only needs 64 cols
  __shared__ float sOutU[MROWS * 128];
  __shared__ float sOutV[MROWS * 128];
  __shared__ float sDiv[MROWS];

  const int tid  = threadIdx.x;
  const int lane = tid & 31;
  const int wid  = tid >> 5;
  const int gs   = blockIdx.x * MROWS;
  const float tval = t[0];

  for (int i = tid; i < MROWS * 64; i += 256)
    sE[i] = eps[(size_t)(gs + (i >> 6)) * 64 + (i & 63)];
  if (tid < MROWS) sDiv[tid] = 0.0f;

  // --- B fragment load from pre-swizzled global (contiguous 32B per lane) ---
  auto bfrag = [&](const bf16* WF, int KT, int nt, int kt) -> v16bf {
    return *(const v16bf*)(WF + (size_t)(((nt * KT) + kt) * 32 + lane) * 16);
  };

  // W2 register-resident: [ts*KT2 + kt], ts0 -> nt=wid, ts1 -> nt=wid+8
  v16bf wB2[2 * KT2];
#pragma unroll
  for (int kt = 0; kt < KT2; ++kt) {
    wB2[kt]       = bfrag(W2F, KT2, wid, kt);
    wB2[KT2 + kt] = bfrag(W2F, KT2, wid + 8, kt);
  }

  // --- A fragment from LDS row-major [16][rs] bf16 (16-bit A 16x32 layout) ---
  auto afrag = [&](const bf16* A, int rs, int kt) -> v16bf {
    int M = lane & 15, half = lane >> 4;
    const bf16* p = A + M * rs + kt * 32 + 16 * half;
    union { v16bf v; u32x4 q[2]; } u;
    u.q[0] = *(const u32x4*)p;
    u.q[1] = *(const u32x4*)(p + 8);
    return u.v;
  };

  const int col  = lane & 15;
  const int half = lane >> 4;

  // build input matrix [MROWS][96]: src 0 = tangent (eps), 1 = [y, t, cond]
  auto build_in = [&](int src, const float* cond, bool bcast) {
    for (int i = tid; i < MROWS * 96; i += 256) {
      int m = i / 96, k = i - m * 96;
      float v = 0.0f;
      if (src == 0) {
        if (k < 64) v = eps[(size_t)(gs + m) * 64 + k];
      } else {
        if (k < 64)       v = x[(size_t)(gs + m) * 65 + k];
        else if (k == 64) v = tval;
        else if (k < 81)  v = bcast ? cond[k - 65]
                                    : cond[(size_t)(gs + m) * 16 + (k - 65)];
      }
      sIn[i] = (bf16)v;
    }
  };

  // 2 M-tiles x 2 N-tiles GEMM, B from registers (layer 2); acc index = mt*2 + ts
  auto gemm4r = [&](const bf16* A, v8f* acc) {
    v8f z = {0.f, 0.f, 0.f, 0.f, 0.f, 0.f, 0.f, 0.f};
    acc[0] = z; acc[1] = z; acc[2] = z; acc[3] = z;
#pragma unroll
    for (int kt = 0; kt < KT2; ++kt) {
      v16bf a0 = afrag(A, 256, kt);
      v16bf a1 = afrag(A + 16 * 256, 256, kt);
      acc[0] = __builtin_amdgcn_wmma_f32_16x16x32_bf16(false, a0, false, wB2[kt],       (short)0, acc[0], false, false);
      acc[1] = __builtin_amdgcn_wmma_f32_16x16x32_bf16(false, a0, false, wB2[KT2 + kt], (short)0, acc[1], false, false);
      acc[2] = __builtin_amdgcn_wmma_f32_16x16x32_bf16(false, a1, false, wB2[kt],       (short)0, acc[2], false, false);
      acc[3] = __builtin_amdgcn_wmma_f32_16x16x32_bf16(false, a1, false, wB2[KT2 + kt], (short)0, acc[3], false, false);
    }
  };
  // 2 M-tiles x 2 N-tiles GEMM, B streamed from global (layer 1)
  auto gemm4g = [&](const bf16* A, int rs, int KT, const bf16* WF, v8f* acc) {
    v8f z = {0.f, 0.f, 0.f, 0.f, 0.f, 0.f, 0.f, 0.f};
    acc[0] = z; acc[1] = z; acc[2] = z; acc[3] = z;
#pragma unroll
    for (int kt = 0; kt < KT; ++kt) {
      v16bf a0 = afrag(A, rs, kt);
      v16bf a1 = afrag(A + 16 * rs, rs, kt);
      v16bf b0 = bfrag(WF, KT, wid, kt);
      v16bf b1v = bfrag(WF, KT, wid + 8, kt);
      acc[0] = __builtin_amdgcn_wmma_f32_16x16x32_bf16(false, a0, false, b0,  (short)0, acc[0], false, false);
      acc[1] = __builtin_amdgcn_wmma_f32_16x16x32_bf16(false, a0, false, b1v, (short)0, acc[1], false, false);
      acc[2] = __builtin_amdgcn_wmma_f32_16x16x32_bf16(false, a1, false, b0,  (short)0, acc[2], false, false);
      acc[3] = __builtin_amdgcn_wmma_f32_16x16x32_bf16(false, a1, false, b1v, (short)0, acc[3], false, false);
    }
  };
  // 2 M-tiles x 1 N-tile GEMM, B streamed from global (layer 3, nt = wid)
  auto gemm2mg = [&](const bf16* A, const bf16* WF, v8f* acc) {
    v8f z = {0.f, 0.f, 0.f, 0.f, 0.f, 0.f, 0.f, 0.f};
    acc[0] = z; acc[1] = z;
#pragma unroll
    for (int kt = 0; kt < KT2; ++kt) {
      v16bf a0 = afrag(A, 256, kt);
      v16bf a1 = afrag(A + 16 * 256, 256, kt);
      v16bf bv = bfrag(WF, KT2, wid, kt);
      acc[0] = __builtin_amdgcn_wmma_f32_16x16x32_bf16(false, a0, false, bv, (short)0, acc[0], false, false);
      acc[1] = __builtin_amdgcn_wmma_f32_16x16x32_bf16(false, a1, false, bv, (short)0, acc[1], false, false);
    }
  };

  // layer 1: mode 0 raw->sDZ1, mode 1 h->sH1, mode 2 h->sH1 + dh->sDH (uses sDZ1)
  auto run_l1 = [&](int mode) {
    v8f acc[4];
    gemm4g(sIn, 96, KT1, W1F, acc);
#pragma unroll
    for (int mt = 0; mt < 2; ++mt) {
#pragma unroll
      for (int ts = 0; ts < 2; ++ts) {
        int nt = ts ? (wid + 8) : wid;
        int nc = nt * 16 + col;
        v8f a = acc[mt * 2 + ts];
#pragma unroll
        for (int r = 0; r < 8; ++r) {
          int idx = (mt * 16 + 8 * half + r) * 256 + nc;
          if (mode == 0) {
            sDZ1[idx] = (bf16)a[r];
          } else {
            float z = a[r] + b1[nc];
            sH1[idx] = (bf16)siluf(z);
            if (mode == 2) sDH[idx] = (bf16)(silup(z) * (float)sDZ1[idx]);
          }
        }
      }
    }
  };

  // layer 2: mode 0 tangent A=sDH -> raw sDZ2; mode 1 fwd A=sH1 -> sDH = silu'(z)*sDZ2;
  //          mode 2 fwd A=sH1 -> sH2 = silu(z)
  auto run_l2 = [&](int mode) {
    const bf16* A = (mode == 0) ? sDH : sH1;
    v8f acc[4];
    gemm4r(A, acc);
#pragma unroll
    for (int mt = 0; mt < 2; ++mt) {
#pragma unroll
      for (int ts = 0; ts < 2; ++ts) {
        int nt = ts ? (wid + 8) : wid;
        int nc = nt * 16 + col;
        v8f a = acc[mt * 2 + ts];
#pragma unroll
        for (int r = 0; r < 8; ++r) {
          int idx = (mt * 16 + 8 * half + r) * 256 + nc;
          if (mode == 0) {
            sDZ2[idx] = (bf16)a[r];
          } else {
            float z = a[r] + b2[nc];
            if (mode == 1) sDH[idx] = (bf16)(silup(z) * (float)sDZ2[idx]);
            else           sH2[idx] = (bf16)siluf(z);
          }
        }
      }
    }
  };

  // layer 3 forward, full N=128 (u, v variants)
  auto run_l3f = [&](float* dst) {
    v8f acc[2];
    gemm2mg(sH2, W3F, acc);
    int nc = wid * 16 + col;
    float bb = b3[nc];
#pragma unroll
    for (int mt = 0; mt < 2; ++mt)
#pragma unroll
      for (int r = 0; r < 8; ++r)
        dst[(mt * 16 + 8 * half + r) * 128 + nc] = acc[mt][r] + bb;
  };

  // layer 3 forward, first 64 cols only (f variant) -> sOutF [MROWS][64]
  auto run_l3f_half = [&]() {
    if (wid < 4) {
      v8f acc[2];
      gemm2mg(sH2, W3F, acc);
      int nc = wid * 16 + col;
      float bb = b3[nc];
#pragma unroll
      for (int mt = 0; mt < 2; ++mt)
#pragma unroll
        for (int r = 0; r < 8; ++r)
          sOutF[(mt * 16 + 8 * half + r) * 64 + nc] = acc[mt][r] + bb;
    }
  };

  // layer 3 tangent: cols 0..63 (W3[:, :64]), waves 0..3; sign * dot(e, dout) -> sDiv
  auto run_l3t = [&](float sign) {
    if (wid < 4) {
      v8f acc[2];
      gemm2mg(sDH, W3F, acc);
      int nc = wid * 16 + col;
#pragma unroll
      for (int mt = 0; mt < 2; ++mt)
#pragma unroll
        for (int r = 0; r < 8; ++r) {
          int M = mt * 16 + 8 * half + r;
          atomicAdd(&sDiv[M], sign * acc[mt][r] * sE[M * 64 + nc]);
        }
    }
  };

  // ---- tangent input: dz1 = e @ W1[:64,:] ----
  build_in(0, nullptr, false); __syncthreads();
  run_l1(0);                   __syncthreads();

  // ---- variant a : cond = cond_u[0] broadcast (negative sign in div) ----
  build_in(1, cond_u, true);   __syncthreads();
  run_l1(2);                   __syncthreads();
  run_l2(0);                   __syncthreads();
  run_l2(1);                   __syncthreads();
  run_l3t(-1.0f);              __syncthreads();

  // ---- variant b : cond = cond_v[0] broadcast (positive sign) ----
  build_in(1, cond_v, true);   __syncthreads();
  run_l1(2);                   __syncthreads();
  run_l2(0);                   __syncthreads();
  run_l2(1);                   __syncthreads();
  run_l3t(1.0f);               __syncthreads();

  // ---- variant f ----
  build_in(1, cond_f, false);  __syncthreads();
  run_l1(1);                   __syncthreads();
  run_l2(2);                   __syncthreads();
  run_l3f_half();              __syncthreads();

  // ---- variant u ----
  build_in(1, cond_u, false);  __syncthreads();
  run_l1(1);                   __syncthreads();
  run_l2(2);                   __syncthreads();
  run_l3f(sOutU);              __syncthreads();

  // ---- variant v ----
  build_in(1, cond_v, false);  __syncthreads();
  run_l1(1);                   __syncthreads();
  run_l2(2);                   __syncthreads();
  run_l3f(sOutV);              __syncthreads();

  // ---- epilogue: out = [ft (64), div + corr_u + corr_v] per sample ----
  for (int i = tid; i < MROWS * 64; i += 256) {
    int m = i >> 6, n = i & 63;
    out[(size_t)(gs + m) * 65 + n] = sOutF[m * 64 + n];
  }
  if (tid < MROWS) {
    int m = tid;
    float cu = 0.0f, cv = 0.0f;
    for (int n = 0; n < 64; ++n) {
      float ft = sOutF[m * 64 + n];
      float ut = sOutU[m * 128 + n];
      float vt = sOutV[m * 128 + n];
      float su = sOutU[m * 128 + 64 + n];
      float sv = sOutV[m * 128 + 64 + n];
      cu += (ft - ut) * su;
      cv += (vt - ft) * sv;
    }
    out[(size_t)(gs + m) * 65 + 64] = sDiv[m] + cu + cv;
  }
}

extern "C" void kernel_launch(void* const* d_in, const int* in_sizes, int n_in,
                              void* d_out, int out_size, void* d_ws, size_t ws_size,
                              hipStream_t stream) {
  (void)n_in; (void)out_size; (void)ws_size;
  const float* t      = (const float*)d_in[0];
  const float* x      = (const float*)d_in[1];
  const float* eps    = (const float*)d_in[2];
  const float* cond_u = (const float*)d_in[3];
  const float* cond_v = (const float*)d_in[4];
  const float* cond_f = (const float*)d_in[5];
  const float* W1     = (const float*)d_in[6];
  const float* b1     = (const float*)d_in[7];
  const float* W2     = (const float*)d_in[8];
  const float* b2     = (const float*)d_in[9];
  const float* W3     = (const float*)d_in[10];
  const float* b3     = (const float*)d_in[11];
  float* out = (float*)d_out;
  bf16* ws   = (bf16*)d_ws;

  const int B = in_sizes[2] / 64;   // eps is (B, 64)

  // 1) convert + pre-swizzle weights into bf16 fragments (L2-resident)
  const int nfrag = NFRAG1 + NFRAG2 + NFRAG3;
  prep_weights<<<(nfrag + 255) / 256, 256, 0, stream>>>(W1, W2, W3, ws);

  // 2) main fused kernel: 32 samples per block, W2 register-resident
  const bf16* W1F = ws;
  const bf16* W2F = ws + W1F_ELEMS;
  const bf16* W3F = ws + W1F_ELEMS + W2F_ELEMS;
  fm_divergence_kernel<<<B / MROWS, 256, 0, stream>>>(
      t, x, eps, cond_u, cond_v, cond_f, b1, b2, b3, W1F, W2F, W3F, out);
}
